// Head_24970939859498
// MI455X (gfx1250) — matrix operand order repaired and checked by reading, not compiled, not run on hardware
//
#include <hip/hip_runtime.h>

// ---------------- problem constants ----------------
#define SEQ   2048
#define EMB   1024
#define HEAD  64
#define BATCH 8
#define MROWS (BATCH * SEQ)   // 16384

typedef __attribute__((ext_vector_type(16))) __bf16        v16bf;
typedef __attribute__((ext_vector_type(8)))  float         v8f;
typedef __attribute__((ext_vector_type(4)))  unsigned int  u32x4;
typedef __attribute__((ext_vector_type(4)))  unsigned int  v4u;
typedef __attribute__((ext_vector_type(4)))  int           v4i;
typedef __attribute__((ext_vector_type(8)))  int           v8i;

// float -> bf16 via native conversion (v_cvt, RNE)
static __device__ __forceinline__ unsigned short f2bf(float f) {
    return __builtin_bit_cast(unsigned short, (__bf16)f);
}

union Frag16 { unsigned short s[16]; u32x4 u[2]; v16bf v; };

// build a 16-element bf16 fragment from two 16B-aligned runs of 8 bf16
static __device__ __forceinline__ v16bf ld2x16B(const unsigned short* p0,
                                                const unsigned short* p1) {
    Frag16 f;
    f.u[0] = *(const u32x4*)p0;
    f.u[1] = *(const u32x4*)p1;
    return f.v;
}

// =====================================================================
// Tensor Data Mover: async-load a 2D bf16 tile [tile_rows x row_elems]
// (row stride = row_elems elements, contiguous rows) from global memory
// into LDS, padding +8 dwords after every 32 dwords so a 64-elem bf16 row
// lands with an 80-element LDS stride (matches fragment-gather layout).
// D# bitfields per CDNA5 ISA ch.8 (group0: count/lds/global/type,
// group1: data_size=2B, pad_enable, pad_interval=32dw, pad_amount=8dw).
// =====================================================================
#if __has_builtin(__builtin_amdgcn_tensor_load_to_lds) && \
    __has_builtin(__builtin_amdgcn_s_wait_tensorcnt)
#define HAVE_TDM 1
#else
#define HAVE_TDM 0
#endif

#if HAVE_TDM
static __device__ __forceinline__ void tdm_load_2d_bf16(
    const void* gptr, unsigned lds_byte_off, int tile_rows, int row_elems)
{
    unsigned long long ga = (unsigned long long)(size_t)gptr;
    v4u g0;
    g0[0] = 1u;                                   // count=1 (valid descriptor)
    g0[1] = lds_byte_off;                         // lds_addr
    g0[2] = (unsigned)(ga & 0xFFFFFFFFu);         // global_addr[31:0]
    g0[3] = (unsigned)((ga >> 32) & 0x01FFFFFFu)  // global_addr[56:32]
          | (2u << 30);                           // type=2 ("image")
    v8i g1;
    g1[0] = (int)((1u << 16)                      // data_size = 2 bytes
                | (1u << 20)                      // pad_enable
                | (4u << 22)                      // pad_interval: 32 dwords
                | (7u << 25));                    // pad_amount : 8 dwords
    g1[1] = (int)((unsigned)row_elems << 16);     // tensor_dim0[15:0]
    g1[2] = 0;                                    // dim0 hi / tensor_dim1 lo
    g1[3] = (int)(16u | ((unsigned)row_elems << 16)); // dim1 hi (=1M rows), tile_dim0
    g1[4] = tile_rows;                            // tile_dim1 (tile_dim2=0)
    g1[5] = row_elems;                            // tensor_dim0_stride[31:0]
    g1[6] = 0;
    g1[7] = 0;
    v4i gz = {0, 0, 0, 0};                        // groups 2/3: dims 2-4 unused
#if __clang_major__ >= 23
    v8i z8 = {0, 0, 0, 0, 0, 0, 0, 0};
    __builtin_amdgcn_tensor_load_to_lds(g0, g1, gz, gz, z8, 0);
#else
    __builtin_amdgcn_tensor_load_to_lds(g0, g1, gz, gz, 0);
#endif
}
#endif

// =====================================================================
// Kernel 1: q/k/v projection.  blockIdx.y selects matrix (0=q,1=k,2=v).
// Block = 256 threads (8 waves); each wave owns 16 rows; block covers 128 rows.
// C = X(16x32 bf16) @ W(32x16 bf16) accumulated over 32 K-chunks.
// 1/sqrt(HEAD)=0.125 folded into q.  Output bf16 into workspace.
// =====================================================================
__global__ __launch_bounds__(256) void qkv_proj_kernel(
    const float* __restrict__ x,  const float* __restrict__ Wk,
    const float* __restrict__ Wq, const float* __restrict__ Wv,
    unsigned short* __restrict__ qkv_ws)
{
    const int y = blockIdx.y;                       // 0=q, 1=k, 2=v
    const float* W = (y == 0) ? Wq : (y == 1) ? Wk : Wv;
    const float scale = (y == 0) ? 0.125f : 1.0f;
    unsigned short* outp = qkv_ws + (size_t)y * MROWS * HEAD;

    // strides padded: 32+8 bf16 = 80B rows (16B aligned)
    __shared__ __align__(16) unsigned short sX [128 * 40]; // x chunk, bf16
    __shared__ __align__(16) unsigned short sWt[ 64 * 40]; // W chunk, transposed

    const int tid  = threadIdx.x;
    const int wave = tid >> 5, lane = tid & 31;
    const int m = lane & 15, h = lane >> 4;         // fragment coords
    const int row0 = blockIdx.x * 128;

    v8f c[4];
    #pragma unroll
    for (int t = 0; t < 4; ++t) c[t] = {};

    for (int kc = 0; kc < EMB / 32; ++kc) {
        __syncthreads();
        // stage X chunk [128 rows x 32 k] -> bf16 (coalesced global reads)
        #pragma unroll
        for (int e = 0; e < 16; ++e) {
            int idx = tid + e * 256;                // 0..4095
            int r = idx >> 5, cc = idx & 31;
            sX[r * 40 + cc] = f2bf(x[(size_t)(row0 + r) * EMB + kc * 32 + cc]);
        }
        // stage W chunk transposed: sWt[col][k]  (contiguous global reads)
        #pragma unroll
        for (int e = 0; e < 8; ++e) {
            int idx = tid + e * 256;                // 0..2047
            int r = idx >> 6, cc = idx & 63;        // r = k row, cc = out col
            sWt[cc * 40 + r] = f2bf(W[(size_t)kc * 2048 + idx]);
        }
        __syncthreads();

        // A fragment (16x32): lane half h covers K = 16h..16h+15 contiguously
        const unsigned short* arow = &sX[(wave * 16 + m) * 40];
        v16bf a = ld2x16B(arow + 16 * h, arow + 16 * h + 8);

        #pragma unroll
        for (int nt = 0; nt < 4; ++nt) {
            // B fragment (32x16): lane half h covers K = 16h..16h+15 contiguously
            const unsigned short* brow = &sWt[(nt * 16 + m) * 40 + 16 * h];
            v16bf bfr = ld2x16B(brow, brow + 8);
            c[nt] = __builtin_amdgcn_wmma_f32_16x16x32_bf16(
                        false, a, false, bfr, (short)0, c[nt], false, false);
        }
    }

    // C layout: vgpr r -> row r + 8h ; lane%16 -> col
    #pragma unroll
    for (int nt = 0; nt < 4; ++nt)
        #pragma unroll
        for (int r = 0; r < 8; ++r) {
            int gr = row0 + wave * 16 + r + 8 * h;
            outp[(size_t)gr * HEAD + nt * 16 + m] = f2bf(c[nt][r] * scale);
        }
}

// =====================================================================
// Kernel 2: fused causal flash-attention over bf16 q/k/v.
// Block = 128 threads (4 waves) = 64 query rows; each wave owns 16 rows.
// Key tiles of 32; online softmax; fp32 accumulators; fp32 output.
// Q/K tiles arrive via the Tensor Data Mover (padded deposit), V is
// transposed manually, P round-trips through per-wave LDS for re-layout.
// =====================================================================
__global__ __launch_bounds__(128) void attn_kernel(
    const unsigned short* __restrict__ qb,
    const unsigned short* __restrict__ kb,
    const unsigned short* __restrict__ vb,
    float* __restrict__ out)
{
    const int b   = blockIdx.y;
    const int qb0 = blockIdx.x * 64;
    const int tid = threadIdx.x;
    const int wave = tid >> 5, lane = tid & 31;
    const int m = lane & 15, h = lane >> 4;

    __shared__ __align__(16) unsigned short sQ [64 * 80]; // 64 q rows x 64 h (+16 pad)
    __shared__ __align__(16) unsigned short sK [32 * 80]; // 32 keys  x 64 h
    __shared__ __align__(16) unsigned short sVt[64 * 40]; // 64 h cols x 32 keys (transposed)
    __shared__ __align__(16) unsigned short sP [64 * 40]; // per-wave 16x32 P tiles

    // ---- stage Q once ----
#if HAVE_TDM
    if (wave == 0) {
        tdm_load_2d_bf16(qb + (size_t)(b * SEQ + qb0) * HEAD,
                         (unsigned)(size_t)(void*)sQ, /*rows=*/64, /*elems=*/64);
        __builtin_amdgcn_s_wait_tensorcnt(0);
    }
#else
    {
        const unsigned int* src =
            (const unsigned int*)(qb + (size_t)(b * SEQ + qb0) * HEAD);
        #pragma unroll
        for (int e = 0; e < 16; ++e) {
            int idx = tid + e * 128;                // dwords 0..2047
            int r = idx >> 5, c2 = idx & 31;
            *(unsigned int*)&sQ[r * 80 + c2 * 2] = src[idx];
        }
    }
#endif
    __syncthreads();

    // Q A-fragments for head-dim chunks 0-31 / 32-63 (persist across key loop)
    const unsigned short* qrow = &sQ[(wave * 16 + m) * 80];
    v16bf aq0 = ld2x16B(qrow +      16 * h, qrow +  8 + 16 * h);
    v16bf aq1 = ld2x16B(qrow + 32 + 16 * h, qrow + 40 + 16 * h);

    float mi[8], li[8];
    v8f o[4];
    #pragma unroll
    for (int r = 0; r < 8; ++r) { mi[r] = -INFINITY; li[r] = 0.0f; }
    #pragma unroll
    for (int t = 0; t < 4; ++t) o[t] = {};

    const int nkt = qb0 / 32 + 2;                   // causal tile limit
    for (int kt = 0; kt < nkt; ++kt) {
        __syncthreads();
        // ---- stage K tile (32x64 bf16) via TDM ----
#if HAVE_TDM
        if (wave == 0) {
            tdm_load_2d_bf16(kb + (size_t)(b * SEQ + kt * 32) * HEAD,
                             (unsigned)(size_t)(void*)sK, /*rows=*/32, /*elems=*/64);
        }
#else
        {
            const unsigned int* src =
                (const unsigned int*)(kb + (size_t)(b * SEQ + kt * 32) * HEAD);
            #pragma unroll
            for (int e = 0; e < 8; ++e) {
                int idx = tid + e * 128;            // dwords 0..1023
                int r = idx >> 5, c2 = idx & 31;
                *(unsigned int*)&sK[r * 80 + c2 * 2] = src[idx];
            }
        }
#endif
        // ---- stage V tile transposed: sVt[headcol][key] (dword reads) ----
        {
            const unsigned int* src =
                (const unsigned int*)(vb + (size_t)(b * SEQ + kt * 32) * HEAD);
            #pragma unroll
            for (int e = 0; e < 8; ++e) {
                int idx = tid + e * 128;            // dwords 0..1023
                int r = idx >> 5, c2 = idx & 31;    // key row, col pair
                unsigned int d = src[idx];
                sVt[(2 * c2    ) * 40 + r] = (unsigned short)(d & 0xFFFFu);
                sVt[(2 * c2 + 1) * 40 + r] = (unsigned short)(d >> 16);
            }
        }
        // prefetch next K/V tile into cache (global_prefetch_b8)
        if (kt + 1 < nkt) {
            size_t noff = (size_t)(b * SEQ + (kt + 1) * 32) * HEAD;
            __builtin_prefetch(kb + noff + tid * 16, 0, 1);
            __builtin_prefetch(vb + noff + tid * 16, 0, 1);
        }
#if HAVE_TDM
        if (wave == 0) __builtin_amdgcn_s_wait_tensorcnt(0);
#endif
        __syncthreads();

        // ---- scores: S(16x32) = Q(16x64) @ K(32x64)^T, two 16x16 frags ----
        v8f s0 = {}, s1 = {};
        {
            const unsigned short* k0 = &sK[m * 80];          // keys 0-15, col = m
            const unsigned short* k1 = &sK[(16 + m) * 80];   // keys 16-31
            v16bf b00 = ld2x16B(k0 + 16 * h,      k0 + 16 * h + 8);  // head 0-31
            v16bf b01 = ld2x16B(k0 + 32 + 16 * h, k0 + 40 + 16 * h); // head 32-63
            v16bf b10 = ld2x16B(k1 + 16 * h,      k1 + 16 * h + 8);
            v16bf b11 = ld2x16B(k1 + 32 + 16 * h, k1 + 40 + 16 * h);
            s0 = __builtin_amdgcn_wmma_f32_16x16x32_bf16(false, aq0, false, b00, (short)0, s0, false, false);
            s0 = __builtin_amdgcn_wmma_f32_16x16x32_bf16(false, aq1, false, b01, (short)0, s0, false, false);
            s1 = __builtin_amdgcn_wmma_f32_16x16x32_bf16(false, aq0, false, b10, (short)0, s1, false, false);
            s1 = __builtin_amdgcn_wmma_f32_16x16x32_bf16(false, aq1, false, b11, (short)0, s1, false, false);
        }

        // ---- causal mask + online softmax (row = r + 8h, col = m / 16+m) ----
        const int kbase = kt * 32;
        float ssc[8];
        #pragma unroll
        for (int r = 0; r < 8; ++r) {
            int qi = qb0 + wave * 16 + r + 8 * h;
            float v0 = (kbase + m      <= qi) ? s0[r] : -INFINITY;
            float v1 = (kbase + 16 + m <= qi) ? s1[r] : -INFINITY;
            float tm = fmaxf(v0, v1);
            #pragma unroll
            for (int d = 1; d < 16; d <<= 1) tm = fmaxf(tm, __shfl_xor(tm, d));
            float mn = fmaxf(mi[r], tm);
            float sc = __expf(mi[r] - mn);          // exp(-inf)=0 on first tile
            float e0 = __expf(v0 - mn);
            float e1 = __expf(v1 - mn);
            float rs = e0 + e1;
            #pragma unroll
            for (int d = 1; d < 16; d <<= 1) rs += __shfl_xor(rs, d);
            li[r] = li[r] * sc + rs;
            mi[r] = mn;
            ssc[r] = sc;
            // scatter P (bf16) to per-wave LDS tile, row-major 16x32
            unsigned short* pr = &sP[(wave * 16 + (r + 8 * h)) * 40];
            pr[m]      = f2bf(e0);
            pr[16 + m] = f2bf(e1);
        }

        // rescale accumulators by exp(m_old - m_new)
        #pragma unroll
        for (int t = 0; t < 4; ++t)
            #pragma unroll
            for (int r = 0; r < 8; ++r) o[t][r] *= ssc[r];

        // intra-wave LDS RAW: P stores -> fragment gather
        asm volatile("s_wait_dscnt 0" ::: "memory");

        // ---- O(16x64) += P(16x32) @ V(32x64) ----
        const unsigned short* prow = &sP[(wave * 16 + m) * 40];
        v16bf ap = ld2x16B(prow + 16 * h, prow + 16 * h + 8);
        #pragma unroll
        for (int t = 0; t < 4; ++t) {
            const unsigned short* vr = &sVt[(t * 16 + m) * 40 + 16 * h];
            v16bf bv = ld2x16B(vr, vr + 8);
            o[t] = __builtin_amdgcn_wmma_f32_16x16x32_bf16(
                       false, ap, false, bv, (short)0, o[t], false, false);
        }
    }

    // ---- epilogue: divide by row sums, write fp32 output ----
    #pragma unroll
    for (int r = 0; r < 8; ++r) {
        int gr = qb0 + wave * 16 + r + 8 * h;
        float inv = 1.0f / li[r];
        size_t base = ((size_t)(b * SEQ) + gr) * HEAD;
        out[base +  0 + m] = o[0][r] * inv;
        out[base + 16 + m] = o[1][r] * inv;
        out[base + 32 + m] = o[2][r] * inv;
        out[base + 48 + m] = o[3][r] * inv;
    }
}

// =====================================================================
extern "C" void kernel_launch(void* const* d_in, const int* in_sizes, int n_in,
                              void* d_out, int out_size, void* d_ws, size_t ws_size,
                              hipStream_t stream) {
    const float* x  = (const float*)d_in[0];
    const float* Wk = (const float*)d_in[1];
    const float* Wq = (const float*)d_in[2];
    const float* Wv = (const float*)d_in[3];

    unsigned short* qkv = (unsigned short*)d_ws;    // 3 * 16384 * 64 bf16 = 6 MB

    dim3 g1(MROWS / 128, 3), b1(256);
    qkv_proj_kernel<<<g1, b1, 0, stream>>>(x, Wk, Wq, Wv, qkv);

    const unsigned short* qb = qkv;
    const unsigned short* kb = qkv + (size_t)MROWS * HEAD;
    const unsigned short* vb = qkv + 2 * (size_t)MROWS * HEAD;

    dim3 g2(SEQ / 64, BATCH), b2(128);
    attn_kernel<<<g2, b2, 0, stream>>>(qb, kb, vb, (float*)d_out);
}
